// Self_attention_block_88940182766146
// MI455X (gfx1250) — matrix operand order repaired and checked by reading, not compiled, not run on hardware
//
#include <hip/hip_runtime.h>
#include <hip/hip_bf16.h>

#define BATCH 8
#define NPTS  2048
#define DPTS  64
#define DM    128
#define KNN   16

typedef __attribute__((ext_vector_type(4)))  float  f32x4;
typedef __attribute__((ext_vector_type(8)))  float  f32x8;
typedef __attribute__((ext_vector_type(16))) __bf16 bf16x16;

__device__ __forceinline__ __bf16 f2bf(float f) {
    union { float f; unsigned u; } a; a.f = f;
    unsigned u = a.u;
    u += 0x7FFFu + ((u >> 16) & 1u);          // round-to-nearest-even
    union { unsigned short s; __bf16 b; } o; o.s = (unsigned short)(u >> 16);
    return o.b;
}
__device__ __forceinline__ float bf2f(__bf16 b) {
    union { __bf16 b; unsigned short s; } i; i.b = b;
    union { unsigned u; float f; } o; o.u = ((unsigned)i.s) << 16;
    return o.f;
}

__device__ __forceinline__ f32x8 wmma_bf16(bf16x16 a, bf16x16 b, f32x8 c) {
    // v_wmma_f32_16x16x32_bf16 : D = A(16x32) * B(32x16) + C(16x16 f32)
    return __builtin_amdgcn_wmma_f32_16x16x32_bf16(false, a, false, b,
                                                   (short)0, c, false, false);
}

// A fragment (16x32 bf16) from row-major [rows][ldk] buffer (LDS or global).
// ISA layout: lanes 0-15 = rows, holding K {k0..k0+7, k0+16..k0+23};
//             lanes 16-31 hold K {k0+8..k0+15, k0+24..k0+31}.
__device__ __forceinline__ bf16x16 ld_a(const __bf16* __restrict__ A, int ldk,
                                        int mt, int k0, int lane) {
    const int r = mt * 16 + (lane & 15);
    const __bf16* p = A + (size_t)r * ldk + k0 + ((lane >> 4) << 3);
    bf16x16 v;
    ((f32x4*)&v)[0] = *(const f32x4*)p;
    ((f32x4*)&v)[1] = *(const f32x4*)(p + 16);
    return v;
}

// B fragment (32x16 bf16) from transposed weight WT[N][ldk] (row n = out-channel,
// contiguous K). lanes 0-15: col n, K {k0..k0+7, k0+16..k0+23}; lanes 16-31: K {k0+8..k0+15, k0+24..k0+31}.
__device__ __forceinline__ bf16x16 ld_b(const __bf16* __restrict__ WT, int ldk,
                                        int nt, int k0, int lane) {
    const __bf16* p = WT + (size_t)(nt * 16 + (lane & 15)) * ldk + k0 + ((lane >> 4) << 3);
    bf16x16 v;
    ((f32x4*)&v)[0] = *(const f32x4*)p;
    ((f32x4*)&v)[1] = *(const f32x4*)(p + 16);
    return v;
}

// One M-tile row of the GEMM: per K-chunk, burst-load ALL NT B fragments into
// distinct registers, then issue NT back-to-back WMMAs on NT accumulators.
// 2*NT loads in flight -> staggered loadcnt waits instead of full drains.
template<int LDA, int LDB, int KC, int NT>
__device__ __forceinline__ void gemm_row(const __bf16* __restrict__ A, int mt,
                                         const __bf16* __restrict__ WT,
                                         int lane, f32x8 (&acc)[NT]) {
    #pragma unroll
    for (int kc = 0; kc < KC; kc++) {
        bf16x16 a = ld_a(A, LDA, mt, kc * 32, lane);
        bf16x16 b[NT];
        #pragma unroll
        for (int nt = 0; nt < NT; nt++) b[nt] = ld_b(WT, LDB, nt, kc * 32, lane);
        #pragma unroll
        for (int nt = 0; nt < NT; nt++) acc[nt] = wmma_bf16(a, b[nt], acc[nt]);
    }
}

// ---------------------------------------------------------------- weights prep
__global__ void transpose_bf16_kernel(const float* __restrict__ in,
                                      __bf16* __restrict__ out, int K, int Nn) {
    int i = blockIdx.x * blockDim.x + threadIdx.x;
    if (i < K * Nn) {
        int k = i / Nn, n = i - k * Nn;
        out[n * K + k] = f2bf(in[i]);      // (in,out) -> (out,in) bf16
    }
}

__global__ void convert_bf16_kernel(const float* __restrict__ in,
                                    __bf16* __restrict__ out, int n) {
    int i = blockIdx.x * blockDim.x + threadIdx.x;
    if (i < n) out[i] = f2bf(in[i]);
}

// ---------------------------------------------------------------- KNN (top-32, stride 2)
__global__ __launch_bounds__(256) void knn_kernel(const float* __restrict__ xyz,
                                                  int* __restrict__ knn) {
    const int bb = blockIdx.y;
    const int pq = blockIdx.x * 256 + threadIdx.x;
    const float* X = xyz + (size_t)bb * NPTS * 3;
    const float qx = X[pq * 3 + 0], qy = X[pq * 3 + 1], qz = X[pq * 3 + 2];
    float bd[32]; int bi[32];
    for (int i = 0; i < 32; i++) { bd[i] = 3.0e38f; bi[i] = 0; }
    __shared__ float sx[256 * 3];
    for (int c0 = 0; c0 < NPTS; c0 += 256) {
        __syncthreads();
        for (int t = threadIdx.x; t < 256 * 3; t += 256) sx[t] = X[(size_t)c0 * 3 + t];
        __syncthreads();
        for (int j = 0; j < 256; j++) {
            const float dx = qx - sx[j * 3], dy = qy - sx[j * 3 + 1], dz = qz - sx[j * 3 + 2];
            const float d = dx * dx + dy * dy + dz * dz;
            if (d < bd[31]) {                 // insertion keeps stable (low-index-first) ties
                int pos = 31;
                while (pos > 0 && bd[pos - 1] > d) {
                    bd[pos] = bd[pos - 1]; bi[pos] = bi[pos - 1]; --pos;
                }
                bd[pos] = d; bi[pos] = c0 + j;
            }
        }
    }
    int* o = knn + ((size_t)bb * NPTS + pq) * KNN;
    for (int i = 0; i < KNN; i++) o[i] = bi[2 * i];   // [:2K:2]
}

// ---------------------------------------------------------------- fc1 + q/k/v
// A fragments for fc1 come straight from global bf16 features (L2-hot).
__global__ __launch_bounds__(128) void qkv_kernel(
    const __bf16* __restrict__ featsb, const float* __restrict__ b1,
    const __bf16* __restrict__ WT1, const __bf16* __restrict__ WTq,
    const __bf16* __restrict__ WTk, const __bf16* __restrict__ WTv,
    float* __restrict__ qg, float* __restrict__ kg, float* __restrict__ vg) {
    __shared__ __attribute__((aligned(16))) __bf16 sX[64 * DM];
    const int row0 = blockIdx.x * 64;
    const int tid = threadIdx.x, lane = tid & 31, w = tid >> 5;
    const int hi = lane >> 4, ln = lane & 15;
    __builtin_prefetch(WT1, 0, 3);
    __builtin_prefetch(WTq, 0, 3);
    const int mt = w;                                   // wave owns M-tile w (its 16 rows)

    // x = feats @ W1 + b1  -> bf16 in sX (own rows only, no barrier needed)
    {
        f32x8 acc[8] = {};
        gemm_row<DPTS, DPTS, 2, 8>(featsb + (size_t)row0 * DPTS, mt, WT1, lane, acc);
        for (int nt = 0; nt < 8; nt++) {
            const int c = nt * 16 + ln;
            const float bv = b1[c];
            for (int j = 0; j < 8; j++)
                sX[(mt * 16 + j + 8 * hi) * DM + c] = f2bf(acc[nt][j] + bv);
        }
    }
    // q/k/v = x @ W
    const __bf16* Ws[3] = { WTq, WTk, WTv };
    float*        Os[3] = { qg,  kg,  vg  };
    for (int t3 = 0; t3 < 3; t3++) {
        f32x8 acc[8] = {};
        gemm_row<DM, DM, 4, 8>(sX, mt, Ws[t3], lane, acc);
        for (int nt = 0; nt < 8; nt++) {
            const int c = nt * 16 + ln;
            for (int j = 0; j < 8; j++)
                Os[t3][(size_t)(row0 + mt * 16 + j + 8 * hi) * DM + c] = acc[nt][j];
        }
    }
}

// ---------------------------------------------------------------- fused attention
// 8 query points / WG -> 128 neighbor rows. 4 waves; wave w owns rows 32w..32w+31
// (tiles 2w, 2w+1 == local points 2w, 2w+1). Fully wave-private -> no barriers.
__global__ __launch_bounds__(128) void attn_kernel(
    const float* __restrict__ xyz, const int* __restrict__ knn,
    const float* __restrict__ qg, const float* __restrict__ kg, const float* __restrict__ vg,
    const float* __restrict__ Wd1, const float* __restrict__ bd1,
    const __bf16* __restrict__ WTd2, const float* __restrict__ bd2,
    const __bf16* __restrict__ WTg1, const float* __restrict__ bg1,
    const __bf16* __restrict__ WTg2, const float* __restrict__ bg2,
    float* __restrict__ attn_out, __bf16* __restrict__ resb) {
    __shared__ __attribute__((aligned(16))) __bf16 sA[128 * DM];   // 32 KB (A operand, reused 3x)
    __shared__ __attribute__((aligned(16))) __bf16 sVP[128 * DM];  // 32 KB (v + pos_enc)
    const int bb = blockIdx.y;
    const int p0 = blockIdx.x * 8;
    const int tid = threadIdx.x, lane = tid & 31, w = tid >> 5;
    const int hi = lane >> 4, ln = lane & 15;
    __builtin_prefetch(WTd2, 0, 3);
    __builtin_prefetch(WTg1, 0, 3);
    __builtin_prefetch(WTg2, 0, 3);

    // each lane owns row r = tid : (point, neighbor) = (tid>>4, tid&15)
    const int mypg = p0 + (tid >> 4);
    const int myidx = knn[((size_t)bb * NPTS + mypg) * KNN + (tid & 15)];
    const float* X = xyz + (size_t)bb * NPTS * 3;
    const float d0 = X[mypg * 3 + 0] - X[myidx * 3 + 0];
    const float d1 = X[mypg * 3 + 1] - X[myidx * 3 + 1];
    const float d2 = X[mypg * 3 + 2] - X[myidx * 3 + 2];

    // phase 1: hid = relu(delta @ Wd1 + bd1) -> sA (bf16, own 32 rows)
    for (int mt2 = 0; mt2 < 2; mt2++) {
        const int mt = 2 * w + mt2;
        for (int nt = 0; nt < 8; nt++) {
            const int c = nt * 16 + ln;
            const float w0 = Wd1[c], w1 = Wd1[DM + c], w2 = Wd1[2 * DM + c], bv = bd1[c];
            for (int j = 0; j < 8; j++) {
                const int sl = mt2 * 16 + j + 8 * hi;        // source lane == row-32w
                const float e0 = __shfl(d0, sl, 32);
                const float e1 = __shfl(d1, sl, 32);
                const float e2 = __shfl(d2, sl, 32);
                sA[(mt * 16 + j + 8 * hi) * DM + c] =
                    f2bf(fmaxf(0.f, e0 * w0 + e1 * w1 + e2 * w2 + bv));
            }
        }
    }

    // phase 2: pos = hid @ Wd2 + bd2   (kept in registers, C-layout)
    f32x8 pos[2][8];
    for (int mt2 = 0; mt2 < 2; mt2++) {
        const int mt = 2 * w + mt2;
        f32x8 acc[8] = {};
        gemm_row<DM, DM, 4, 8>(sA, mt, WTd2, lane, acc);
        for (int nt = 0; nt < 8; nt++) {
            const float bv = bd2[nt * 16 + ln];
            for (int j = 0; j < 8; j++) acc[nt][j] += bv;
            pos[mt2][nt] = acc[nt];
        }
    }

    // phase 3: h = q - k + pos -> sA ; vp = v + pos -> sVP  (k/v gathers, L2-hot)
    const size_t fbase = (size_t)bb * NPTS * DM;
    for (int mt2 = 0; mt2 < 2; mt2++) {
        const int mt = 2 * w + mt2;
        const int pg = p0 + mt;
        for (int nt = 0; nt < 8; nt++) {
            const int c = nt * 16 + ln;
            const float qv = qg[fbase + (size_t)pg * DM + c];
            for (int j = 0; j < 8; j++) {
                const int sl = mt2 * 16 + j + 8 * hi;
                const int idx = __shfl(myidx, sl, 32);
                const float kv = kg[fbase + (size_t)idx * DM + c];
                const float vv = vg[fbase + (size_t)idx * DM + c];
                const float pv = pos[mt2][nt][j];
                const int r = mt * 16 + j + 8 * hi;
                sA[r * DM + c]  = f2bf(qv - kv + pv);
                sVP[r * DM + c] = f2bf(vv + pv);
            }
        }
    }

    const float scale = 0.088388347648318447f;   // 1/sqrt(128)
    for (int mt2 = 0; mt2 < 2; mt2++) {
        const int mt = 2 * w + mt2;
        const int pg = p0 + mt;
        // phase 4: g1 = relu(h @ Wg1 + bg1) -> sA  (all reads of tile mt before writes)
        f32x8 g[8] = {};
        gemm_row<DM, DM, 4, 8>(sA, mt, WTg1, lane, g);
        for (int nt = 0; nt < 8; nt++) {
            const int c = nt * 16 + ln;
            const float bv = bg1[c];
            for (int j = 0; j < 8; j++)
                sA[(mt * 16 + j + 8 * hi) * DM + c] = f2bf(fmaxf(0.f, g[nt][j] + bv));
        }
        // phase 5: logits = g1 @ Wg2 + bg2
        #pragma unroll
        for (int nt = 0; nt < 8; nt++) g[nt] = f32x8{0.f, 0.f, 0.f, 0.f, 0.f, 0.f, 0.f, 0.f};
        gemm_row<DM, DM, 4, 8>(sA, mt, WTg2, lane, g);
        // phase 6/7: per-channel softmax over 16 neighbors + weighted sum
        for (int nt = 0; nt < 8; nt++) {
            const int c = nt * 16 + ln;
            const float bv = bg2[c];
            float x[8], e[8];
            float mx = -3.4e38f;
            for (int j = 0; j < 8; j++) { x[j] = (g[nt][j] + bv) * scale; mx = fmaxf(mx, x[j]); }
            mx = fmaxf(mx, __shfl_xor(mx, 16, 32));       // other half of the 16 rows
            float s = 0.f;
            for (int j = 0; j < 8; j++) { e[j] = __expf(x[j] - mx); s += e[j]; }
            s += __shfl_xor(s, 16, 32);
            const float inv = 1.f / s;
            float part = 0.f;
            for (int j = 0; j < 8; j++) {
                const int m = j + 8 * hi;
                const float a = e[j] * inv;
                attn_out[(((size_t)bb * NPTS + pg) * KNN + m) * DM + c] = a;
                part += a * bf2f(sVP[(mt * 16 + m) * DM + c]);
            }
            part += __shfl_xor(part, 16, 32);
            if (hi == 0) resb[((size_t)bb * NPTS + pg) * DM + c] = f2bf(part);
        }
    }
}

// ---------------------------------------------------------------- fc2 + residual
// A fragments straight from global bf16 result buffer: no LDS, no barriers.
__global__ __launch_bounds__(128) void fc2_kernel(
    const __bf16* __restrict__ resb, const __bf16* __restrict__ WT2,
    const float* __restrict__ b2, const float* __restrict__ feats,
    float* __restrict__ out) {
    const int row0 = blockIdx.x * 64;
    const int tid = threadIdx.x, lane = tid & 31, w = tid >> 5;
    const int hi = lane >> 4, ln = lane & 15;
    const int mt = w;
    f32x8 acc[4] = {};
    gemm_row<DM, DM, 4, 4>(resb + (size_t)row0 * DM, mt, WT2, lane, acc);
    for (int nt = 0; nt < 4; nt++) {
        const int c = nt * 16 + ln;
        const float bv = b2[c];
        for (int j = 0; j < 8; j++) {
            const size_t r = row0 + mt * 16 + j + 8 * hi;
            out[r * DPTS + c] = acc[nt][j] + bv + feats[r * DPTS + c];
        }
    }
}

// ---------------------------------------------------------------- launch
extern "C" void kernel_launch(void* const* d_in, const int* in_sizes, int n_in,
                              void* d_out, int out_size, void* d_ws, size_t ws_size,
                              hipStream_t stream) {
    (void)in_sizes; (void)n_in; (void)out_size; (void)ws_size;
    const float* xyz   = (const float*)d_in[0];
    const float* feats = (const float*)d_in[1];
    const float* W1 = (const float*)d_in[2];  const float* b1 = (const float*)d_in[3];
    const float* W2 = (const float*)d_in[4];  const float* b2 = (const float*)d_in[5];
    const float* Wd1 = (const float*)d_in[6]; const float* bd1 = (const float*)d_in[7];
    const float* Wd2 = (const float*)d_in[8]; const float* bd2 = (const float*)d_in[9];
    const float* Wg1 = (const float*)d_in[10]; const float* bg1 = (const float*)d_in[11];
    const float* Wg2 = (const float*)d_in[12]; const float* bg2 = (const float*)d_in[13];
    const float* Wq = (const float*)d_in[14];
    const float* Wk = (const float*)d_in[15];
    const float* Wv = (const float*)d_in[16];

    char* p = (char*)d_ws;
    int*    knn    = (int*)p;    p += (size_t)BATCH * NPTS * KNN * 4;
    float*  qg     = (float*)p;  p += (size_t)BATCH * NPTS * DM * 4;
    float*  kg     = (float*)p;  p += (size_t)BATCH * NPTS * DM * 4;
    float*  vg     = (float*)p;  p += (size_t)BATCH * NPTS * DM * 4;
    __bf16* resb   = (__bf16*)p; p += (size_t)BATCH * NPTS * DM * 2;
    __bf16* featsb = (__bf16*)p; p += (size_t)BATCH * NPTS * DPTS * 2;
    __bf16* WT1  = (__bf16*)p; p += (size_t)DPTS * DM * 2;
    __bf16* WTq  = (__bf16*)p; p += (size_t)DM * DM * 2;
    __bf16* WTk  = (__bf16*)p; p += (size_t)DM * DM * 2;
    __bf16* WTv  = (__bf16*)p; p += (size_t)DM * DM * 2;
    __bf16* WTd2 = (__bf16*)p; p += (size_t)DM * DM * 2;
    __bf16* WTg1 = (__bf16*)p; p += (size_t)DM * DM * 2;
    __bf16* WTg2 = (__bf16*)p; p += (size_t)DM * DM * 2;
    __bf16* WT2  = (__bf16*)p; p += (size_t)DM * DPTS * 2;

    float* out_res  = (float*)d_out;
    float* out_attn = out_res + (size_t)BATCH * NPTS * DPTS;

    dim3 tb(256);
    transpose_bf16_kernel<<<dim3((DPTS * DM + 255) / 256), tb, 0, stream>>>(W1, WT1, DPTS, DM);
    transpose_bf16_kernel<<<dim3((DM * DM + 255) / 256), tb, 0, stream>>>(Wq, WTq, DM, DM);
    transpose_bf16_kernel<<<dim3((DM * DM + 255) / 256), tb, 0, stream>>>(Wk, WTk, DM, DM);
    transpose_bf16_kernel<<<dim3((DM * DM + 255) / 256), tb, 0, stream>>>(Wv, WTv, DM, DM);
    transpose_bf16_kernel<<<dim3((DM * DM + 255) / 256), tb, 0, stream>>>(Wd2, WTd2, DM, DM);
    transpose_bf16_kernel<<<dim3((DM * DM + 255) / 256), tb, 0, stream>>>(Wg1, WTg1, DM, DM);
    transpose_bf16_kernel<<<dim3((DM * DM + 255) / 256), tb, 0, stream>>>(Wg2, WTg2, DM, DM);
    transpose_bf16_kernel<<<dim3((DM * DPTS + 255) / 256), tb, 0, stream>>>(W2, WT2, DM, DPTS);
    convert_bf16_kernel<<<dim3((BATCH * NPTS * DPTS + 255) / 256), tb, 0, stream>>>(
        feats, featsb, BATCH * NPTS * DPTS);

    knn_kernel<<<dim3(NPTS / 256, BATCH), dim3(256), 0, stream>>>(xyz, knn);
    qkv_kernel<<<dim3(BATCH * NPTS / 64), dim3(128), 0, stream>>>(
        featsb, b1, WT1, WTq, WTk, WTv, qg, kg, vg);
    attn_kernel<<<dim3(NPTS / 8, BATCH), dim3(128), 0, stream>>>(
        xyz, knn, qg, kg, vg, Wd1, bd1, WTd2, bd2, WTg1, bg1, WTg2, bg2, out_attn, resb);
    fc2_kernel<<<dim3(BATCH * NPTS / 64), dim3(128), 0, stream>>>(resb, WT2, b2, feats, out_res);
}
